// BoardLoss_11793980194936
// MI455X (gfx1250) — compile-verified, matrix operand order checked by reading, not run on hardware
//
#include <hip/hip_runtime.h>

// BoardLoss on MI455X (gfx1250, wave32).
// Memory-bound single pass over 256MB fp32 -> scalar. HBM floor ~11us @23.3TB/s.
// WMMA (f32_16x16x32_f16, B=ones) performs the cross-wave column-count reduction.

typedef __attribute__((ext_vector_type(16))) _Float16 v16h;
typedef __attribute__((ext_vector_type(8)))  float    v8f;

static constexpr int kS     = 512;   // board side
static constexpr int kWaves = 16;    // waves per block
static constexpr int kTPB   = kWaves * 32;

__global__ __launch_bounds__(kTPB)
void board_loss_batch(const float* __restrict__ x, float* __restrict__ ws,
                      float invBS, float inv6B) {
  const int tid  = threadIdx.x;
  const int wave = tid >> 5;
  const int lane = tid & 31;
  const float* __restrict__ xb = x + (size_t)blockIdx.x * (kS * kS);

  __shared__ _Float16 colf[kWaves][kS];   // per-wave column counts (exact in f16)
  __shared__ float wrow[kWaves];
  __shared__ float wcol[kWaves];
  __shared__ int   wrun[kWaves];

  float rowloss = 0.f;
  int   runcnt  = 0;
  unsigned cp0 = 0, cp1 = 0, cp2 = 0, cp3 = 0;   // 16 packed byte counters

  // lane 31 holds columns 496..511: triple starts only valid up to col 509 (p<=13)
  const unsigned validMask = (lane == 31) ? 0x3FFFu : 0xFFFFu;

  for (int r = wave; r < kS; r += kWaves) {
    const float4* row4 = (const float4*)(xb + r * kS + lane * 16);
    if (r + kWaves < kS)
      __builtin_prefetch((const void*)(xb + (r + kWaves) * kS + lane * 16), 0, 0);

    float4 a0 = row4[0], a1 = row4[1], a2 = row4[2], a3 = row4[3];
    unsigned m = 0;
    m |= (a0.x > 0.5f) ? 1u << 0  : 0u;  m |= (a0.y > 0.5f) ? 1u << 1  : 0u;
    m |= (a0.z > 0.5f) ? 1u << 2  : 0u;  m |= (a0.w > 0.5f) ? 1u << 3  : 0u;
    m |= (a1.x > 0.5f) ? 1u << 4  : 0u;  m |= (a1.y > 0.5f) ? 1u << 5  : 0u;
    m |= (a1.z > 0.5f) ? 1u << 6  : 0u;  m |= (a1.w > 0.5f) ? 1u << 7  : 0u;
    m |= (a2.x > 0.5f) ? 1u << 8  : 0u;  m |= (a2.y > 0.5f) ? 1u << 9  : 0u;
    m |= (a2.z > 0.5f) ? 1u << 10 : 0u;  m |= (a2.w > 0.5f) ? 1u << 11 : 0u;
    m |= (a3.x > 0.5f) ? 1u << 12 : 0u;  m |= (a3.y > 0.5f) ? 1u << 13 : 0u;
    m |= (a3.z > 0.5f) ? 1u << 14 : 0u;  m |= (a3.w > 0.5f) ? 1u << 15 : 0u;

    // column counters: spread 4 bits to 4 bytes and accumulate (max 32, no overflow)
    cp0 += (( m         & 0xFu) * 0x00204081u) & 0x01010101u;
    cp1 += (((m >> 4)   & 0xFu) * 0x00204081u) & 0x01010101u;
    cp2 += (((m >> 8)   & 0xFu) * 0x00204081u) & 0x01010101u;
    cp3 += (((m >> 12)  & 0xFu) * 0x00204081u) & 0x01010101u;

    // row sum (wave32 reduce of popcounts)
    int rs = __popc(m);
    rs += __shfl_xor(rs, 16, 32);
    rs += __shfl_xor(rs, 8, 32);
    rs += __shfl_xor(rs, 4, 32);
    rs += __shfl_xor(rs, 2, 32);
    rs += __shfl_xor(rs, 1, 32);
    float dv = (float)rs - 3.0f;
    rowloss += dv * dv;

    // run-of-3 detection along the row (needs next lane's low 2 bits)
    unsigned next2 = __shfl_down(m, 1, 32) & 3u;
    unsigned v   = m | (next2 << 16);          // 18 significant bits
    unsigned eq  = ~(v ^ (v >> 1));            // bit p: b[p]==b[p+1]
    unsigned r3  = eq & (eq >> 1);             // bit p: run of 3 starting at p
    int hit = (r3 & validMask) != 0u;
    runcnt += __any(hit) ? 1 : 0;              // uniform across wave
  }

  // publish per-wave column counters (exact small ints) as f16 for WMMA
  {
    const int base = lane * 16;
    unsigned packs[4] = {cp0, cp1, cp2, cp3};
#pragma unroll
    for (int q = 0; q < 4; ++q) {
#pragma unroll
      for (int j = 0; j < 4; ++j) {
        colf[wave][base + q * 4 + j] =
            (_Float16)(float)((packs[q] >> (8 * j)) & 0xFFu);
      }
    }
  }
  if (lane == 0) { wrow[wave] = rowloss; wrun[wave] = runcnt; }
  __syncthreads();

  // Cross-wave column reduction via V_WMMA_F32_16X16X32_F16 with B = ones.
  // A-layout (documented): lanes 0-15 hold M=0..15 / K=0..7 (+K=16..23 zero),
  // lanes 16-31 hold M=0..15 / K=8..15 (+K=24..31 zero). K indexes the wave.
  const int ml = lane & 15;
  const int kb = (lane < 16) ? 0 : 8;

  v16h ones;
#pragma unroll
  for (int j = 0; j < 16; ++j) ones[j] = (_Float16)1.0f;

  float cl = 0.f;
#pragma unroll
  for (int ti = 0; ti < 2; ++ti) {
    const int t = wave + ti * kWaves;          // 32 tiles of 16 columns
    const int c = t * 16 + ml;
    v16h a;
#pragma unroll
    for (int j = 0; j < 8; ++j) a[j] = colf[kb + j][c];
#pragma unroll
    for (int j = 8; j < 16; ++j) a[j] = (_Float16)0.0f;
    v8f c0 = {};
    v8f d = __builtin_amdgcn_wmma_f32_16x16x32_f16(
        /*neg_a=*/false, a, /*neg_b=*/false, ones,
        /*c_mod=*/(short)0, c0, /*reuse_a=*/false, /*reuse_b=*/false);
    // D layout: lane 0 VGPR p = colsum(16t+p); lane 16 VGPR p = colsum(16t+8+p)
#pragma unroll
    for (int p = 0; p < 8; ++p) { float e = d[p] - 3.0f; cl += e * e; }
  }
  float clm = (lane == 0 || lane == 16) ? cl : 0.f;   // rows 0..7 + 8..15 per tile
  clm += __shfl_xor(clm, 16, 32);
  clm += __shfl_xor(clm, 8, 32);
  clm += __shfl_xor(clm, 4, 32);
  clm += __shfl_xor(clm, 2, 32);
  clm += __shfl_xor(clm, 1, 32);
  if (lane == 0) wcol[wave] = clm;
  __syncthreads();

  if (tid == 0) {
    float rsum = 0.f, csum = 0.f; int rn = 0;
    for (int w = 0; w < kWaves; ++w) { rsum += wrow[w]; csum += wcol[w]; rn += wrun[w]; }
    ws[blockIdx.x] = (rsum + csum) * invBS + (float)rn * inv6B;
  }
}

__global__ __launch_bounds__(256)
void board_loss_finalize(const float* __restrict__ ws, float* __restrict__ out, int n) {
  const int t = threadIdx.x;
  float v = 0.f;
  for (int i = t; i < n; i += 256) v += ws[i];
  v += __shfl_xor(v, 16, 32);
  v += __shfl_xor(v, 8, 32);
  v += __shfl_xor(v, 4, 32);
  v += __shfl_xor(v, 2, 32);
  v += __shfl_xor(v, 1, 32);
  __shared__ float s[8];
  if ((t & 31) == 0) s[t >> 5] = v;
  __syncthreads();
  if (t == 0) {
    float tot = 0.f;
    for (int i = 0; i < 8; ++i) tot += s[i];
    out[0] = tot;
  }
}

extern "C" void kernel_launch(void* const* d_in, const int* in_sizes, int n_in,
                              void* d_out, int out_size, void* d_ws, size_t ws_size,
                              hipStream_t stream) {
  const float* x = (const float*)d_in[0];
  float* out = (float*)d_out;
  float* ws  = (float*)d_ws;

  const int B = in_sizes[0] / (kS * kS);          // 256
  const float invBS = 1.0f / ((float)B * (float)kS);
  const float inv6B = 1.0f / (6.0f * (float)B);

  board_loss_batch<<<B, kTPB, 0, stream>>>(x, ws, invBS, inv6B);
  board_loss_finalize<<<1, 256, 0, stream>>>(ws, out, B);
}
